// Decoder_49426483642914
// MI455X (gfx1250) — compile-verified, matrix operand order checked by reading, not run on hardware
//
#include <hip/hip_runtime.h>
#include <hip/hip_bf16.h>

// ---------------------------------------------------------------------------
// Problem constants (from reference)
// ---------------------------------------------------------------------------
#define BB   64
#define PP   196
#define ENCD 2048
#define AD   512
#define HD   512
#define ED   512
#define VV   30000
#define LL   30
#define TT   29          // L - 1 decode steps
#define XD   2560        // E + ENC (LSTM input width)
#define GD   2048        // 4*H gate width

typedef __attribute__((ext_vector_type(16))) __bf16 v16bf;
typedef __attribute__((ext_vector_type(8)))  __bf16 v8bf;
typedef __attribute__((ext_vector_type(8)))  float  v8f;

// fp32 -> bf16 round-to-nearest-even
__device__ __forceinline__ __bf16 f2bf(float f) {
    unsigned u = __builtin_bit_cast(unsigned, f);
    unsigned r = (u + 0x7FFFu + ((u >> 16) & 1u)) >> 16;
    unsigned short s = (unsigned short)r;
    return __builtin_bit_cast(__bf16, s);
}

__device__ __forceinline__ float sigf(float x) { return 1.0f / (1.0f + __expf(-x)); }

// Load one 16-element bf16 fragment: K-packs [0..7] and [16..23] relative to p.
// p is 16-byte aligned -> two global_load_b128.
__device__ __forceinline__ v16bf ldfrag(const __bf16* __restrict__ p) {
    v8bf lo = *(const v8bf*)(p);
    v8bf hi = *(const v8bf*)(p + 16);
    return __builtin_shufflevector(lo, hi, 0, 1, 2, 3, 4, 5, 6, 7,
                                           8, 9, 10, 11, 12, 13, 14, 15);
}

// ---------------------------------------------------------------------------
// WMMA GEMM:  C[M x N] = A[M x K](bf16) * W[N x K](bf16)^T + bias  (f32 accum)
//   * both operands row-major, K-contiguous -> pure b128 fragment loads
//   * one wave computes a 64(M) x 32(N) slab: 4 M-tiles x 2 N-tiles
//     -> per K-step: 4 A-frags + 2 B-frags feed 8 WMMAs (1.5 b128/WMMA)
//   * 4 waves / 128-thread block -> block covers 128 columns
//   * requires M % 64 == 0, N % 16 == 0, K % 32 == 0 (true for all call sites)
// ---------------------------------------------------------------------------
__global__ void __launch_bounds__(128)
gemm_bf16(const __bf16* __restrict__ A, int lda,
          const __bf16* __restrict__ W, int ldw,
          const float* __restrict__ bias,
          float* __restrict__ C, int ldc,
          int M, int N, int K, int accFlag)
{
    const int lane   = threadIdx.x & 31;
    const int wave   = threadIdx.x >> 5;
    const int laneLo = lane & 15;
    const int laneHi = lane >> 4;
    const int n0 = (blockIdx.x * 4 + wave) * 32;
    const int m0 = blockIdx.y * 64;
    if (__builtin_amdgcn_readfirstlane(n0 >= N ? 1 : 0)) return;   // scalar branch
    const bool two = __builtin_amdgcn_readfirstlane((n0 + 16) < N ? 1 : 0) != 0;

    // ISA 16-bit A layout: lane holds row M=laneLo; K packs 8+8 shifted by laneHi*8
    const __bf16* Ap[4];
    #pragma unroll
    for (int r = 0; r < 4; ++r)
        Ap[r] = A + (size_t)(m0 + r * 16 + laneLo) * lda + laneHi * 8;
    const __bf16* Wp0 = W + (size_t)(n0 + laneLo) * ldw + laneHi * 8;
    const __bf16* Wp1 = Wp0 + (size_t)16 * ldw;

    v8f acc0[4] = {};
    v8f acc1[4] = {};
    for (int k = 0; k < K; k += 32) {
        v16bf b0 = ldfrag(Wp0 + k);
        __builtin_prefetch((const void*)(Wp0 + k + 256), 0, 1);  // global_prefetch_b8
        v16bf ar[4];
        #pragma unroll
        for (int r = 0; r < 4; ++r) ar[r] = ldfrag(Ap[r] + k);
        #pragma unroll
        for (int r = 0; r < 4; ++r)
            acc0[r] = __builtin_amdgcn_wmma_f32_16x16x32_bf16(
                          false, ar[r], false, b0, (short)0, acc0[r], false, false);
        if (two) {
            v16bf b1 = ldfrag(Wp1 + k);
            #pragma unroll
            for (int r = 0; r < 4; ++r)
                acc1[r] = __builtin_amdgcn_wmma_f32_16x16x32_bf16(
                              false, ar[r], false, b1, (short)0, acc1[r], false, false);
        }
    }

    const float bn0 = bias ? bias[n0 + laneLo] : 0.0f;
    #pragma unroll
    for (int r = 0; r < 4; ++r) {
        #pragma unroll
        for (int i = 0; i < 8; ++i) {           // C/D layout: vgpr i -> row i + 8*laneHi
            const int m = m0 + r * 16 + i + laneHi * 8;
            const size_t idx = (size_t)m * ldc + (n0 + laneLo);
            float v = acc0[r][i] + bn0;
            if (accFlag) v += C[idx];
            C[idx] = v;
        }
    }
    if (two) {
        const float bn1 = bias ? bias[n0 + 16 + laneLo] : 0.0f;
        #pragma unroll
        for (int r = 0; r < 4; ++r) {
            #pragma unroll
            for (int i = 0; i < 8; ++i) {
                const int m = m0 + r * 16 + i + laneHi * 8;
                const size_t idx = (size_t)m * ldc + (n0 + 16 + laneLo);
                float v = acc1[r][i] + bn1;
                if (accFlag) v += C[idx];
                C[idx] = v;
            }
        }
    }
}

// ---------------------------------------------------------------------------
// Small helper kernels
// ---------------------------------------------------------------------------

// Rank-sort 64 caption lengths (descending, stable) -> sort_ind, dec_len
__global__ void sort_kernel(const int* __restrict__ lengths, int* __restrict__ sort_ind,
                            int* __restrict__ dec_len)
{
    __shared__ int len[BB];
    int i = threadIdx.x;
    len[i] = lengths[i];            // caption_lengths is (B,1)
    __syncthreads();
    int li = len[i], rank = 0;
    for (int j = 0; j < BB; ++j) {
        int lj = len[j];
        if (lj > li || (lj == li && j < i)) rank++;
    }
    sort_ind[rank] = i;
    dec_len[rank]  = li - 1;
}

// caps / dec_len / sort_ind -> float outputs
__global__ void meta_out_kernel(const int* __restrict__ sort_ind, const int* __restrict__ dec_len,
                                const int* __restrict__ caps,
                                float* __restrict__ out_caps, float* __restrict__ out_declen,
                                float* __restrict__ out_sortind)
{
    int i = blockIdx.x * 256 + threadIdx.x;
    if (i < BB * LL) {
        int b = i / LL, l = i - b * LL;
        out_caps[i] = (float)caps[sort_ind[b] * LL + l];
    }
    if (i < BB) {
        out_declen[i]  = (float)dec_len[i];
        out_sortind[i] = (float)sort_ind[i];
    }
}

// fp32 -> bf16 conversion (weights and small activations)
__global__ void f2bf_kernel(const float* __restrict__ src, __bf16* __restrict__ dst, size_t n)
{
    size_t i = (size_t)blockIdx.x * blockDim.x + threadIdx.x;
    if (i < n) dst[i] = f2bf(src[i]);
}

// gather encoder rows through sort_ind and convert to bf16: enc_bf[(b,p),d]
__global__ void enc_gather_bf16_kernel(const float* __restrict__ enc,
                                       const int* __restrict__ sort_ind,
                                       __bf16* __restrict__ dst)
{
    int b = blockIdx.y;
    int i = blockIdx.x * 256 + threadIdx.x;          // 0 .. PP*ENCD-1
    const float* src = enc + (size_t)sort_ind[b] * PP * ENCD;
    dst[(size_t)b * PP * ENCD + i] = f2bf(src[i]);
}

// mean over P of gathered encoder rows -> mean_enc bf16 (B x ENC)
__global__ void mean_kernel(const float* __restrict__ enc, const int* __restrict__ sort_ind,
                            __bf16* __restrict__ mean_enc)
{
    int b = blockIdx.y;
    int d = blockIdx.x * 256 + threadIdx.x;
    const float* src = enc + (size_t)sort_ind[b] * PP * ENCD + d;
    float s = 0.f;
    for (int p = 0; p < PP; ++p) s += src[(size_t)p * ENCD];
    mean_enc[(size_t)b * ENCD + d] = f2bf(s * (1.0f / (float)PP));
}

// e[b,p] = full_att_b + sum_a relu(att1[b,p,a] + att2[b,a]) * full_att_w[a]  (one wave per (b,p))
__global__ void att_e_kernel(const float* __restrict__ att1, const float* __restrict__ att2,
                             const float* __restrict__ faw, const float* __restrict__ fab,
                             float* __restrict__ e)
{
    int gw   = (int)((blockIdx.x * blockDim.x + threadIdx.x) >> 5);
    int lane = threadIdx.x & 31;
    if (gw >= BB * PP) return;
    int b = gw / PP;
    const float* a1 = att1 + (size_t)gw * AD;
    const float* a2 = att2 + (size_t)b  * AD;
    float s = 0.f;
    for (int a = lane; a < AD; a += 32) {
        float v = a1[a] + a2[a];
        v = v > 0.f ? v : 0.f;
        s += v * faw[a];
    }
    #pragma unroll
    for (int off = 16; off > 0; off >>= 1) s += __shfl_down(s, off, 32);
    if (lane == 0) e[gw] = s + fab[0];
}

// softmax over P per batch row; also writes masked alphas into output
__global__ void softmax_kernel(const float* __restrict__ e, float* __restrict__ alpha,
                               float* __restrict__ out_alphas, const int* __restrict__ dec_len,
                               int t)
{
    __shared__ float red[256];
    int b = blockIdx.x, tid = threadIdx.x;
    float v = (tid < PP) ? e[b * PP + tid] : -3.4e38f;
    red[tid] = v;  __syncthreads();
    for (int s = 128; s > 0; s >>= 1) { if (tid < s) red[tid] = fmaxf(red[tid], red[tid + s]); __syncthreads(); }
    float mx = red[0];  __syncthreads();
    float ex = (tid < PP) ? __expf(v - mx) : 0.f;
    red[tid] = ex;  __syncthreads();
    for (int s = 128; s > 0; s >>= 1) { if (tid < s) red[tid] += red[tid + s]; __syncthreads(); }
    float inv = 1.0f / red[0];
    if (tid < PP) {
        float a = ex * inv;
        alpha[b * PP + tid] = a;
        float mask = (t < dec_len[b]) ? 1.0f : 0.0f;
        out_alphas[((size_t)b * TT + t) * PP + tid] = a * mask;
    }
}

// awe[b,d] = sum_p alpha[b,p] * enc[sort_ind[b],p,d]   (fp32 for accuracy)
__global__ void awe_kernel(const float* __restrict__ alpha, const float* __restrict__ enc,
                           const int* __restrict__ sort_ind, float* __restrict__ awe)
{
    __shared__ float al[PP];
    int b = blockIdx.y;
    int d = blockIdx.x * 256 + threadIdx.x;
    for (int p = threadIdx.x; p < PP; p += 256) al[p] = alpha[b * PP + p];
    __syncthreads();
    const float* src = enc + (size_t)sort_ind[b] * PP * ENCD + d;
    float s = 0.f;
    for (int p = 0; p < PP; ++p) s += al[p] * src[(size_t)p * ENCD];
    awe[(size_t)b * ENCD + d] = s;
}

// x_bf[b,0:512] = bf16(embed[caps[b,t]]);  x_bf[b,512:2560] = bf16(sigmoid(gate_lin)*awe)
__global__ void gate_x_kernel(const float* __restrict__ gate_lin, const float* __restrict__ awe,
                              const float* __restrict__ embed, const int* __restrict__ caps,
                              const int* __restrict__ sort_ind, int t, __bf16* __restrict__ x)
{
    int b = blockIdx.y;
    int i = blockIdx.x * 256 + threadIdx.x;          // 0..2559
    float v;
    if (i < ED) {
        int cap = caps[sort_ind[b] * LL + t];
        v = embed[(size_t)cap * ED + i];
    } else {
        int d = i - ED;
        float g = sigf(gate_lin[(size_t)b * ENCD + d]);
        v = g * awe[(size_t)b * ENCD + d];
    }
    x[(size_t)b * XD + i] = f2bf(v);
}

// LSTM pointwise: gates (B x 4H, order i,f,g,o) + c -> new c, h (fp32) and h_bf (bf16)
__global__ void lstm_kernel(const float* __restrict__ gates, float* __restrict__ c,
                            float* __restrict__ h, __bf16* __restrict__ h_bf)
{
    int b = blockIdx.y;
    int j = blockIdx.x * 256 + threadIdx.x;          // 0..511
    const float* g = gates + (size_t)b * GD;
    float ig = sigf(g[j]);
    float fg = sigf(g[HD + j]);
    float gg = tanhf(g[2 * HD + j]);
    float og = sigf(g[3 * HD + j]);
    float cn = fg * c[(size_t)b * HD + j] + ig * gg;
    float hn = og * tanhf(cn);
    c[(size_t)b * HD + j]    = cn;
    h[(size_t)b * HD + j]    = hn;
    h_bf[(size_t)b * HD + j] = f2bf(hn);
}

// masked scatter of step predictions into output (B,T,V)
__global__ void preds_out_kernel(const float* __restrict__ preds, const int* __restrict__ dec_len,
                                 int t, float* __restrict__ out)
{
    int b = blockIdx.y;
    int v = blockIdx.x * 256 + threadIdx.x;
    if (v >= VV) return;
    float mask = (t < dec_len[b]) ? 1.0f : 0.0f;
    out[((size_t)b * TT + t) * VV + v] = preds[(size_t)b * VV + v] * mask;
}

// ---------------------------------------------------------------------------
// Launch
// ---------------------------------------------------------------------------
static inline void launch_gemm(hipStream_t st, const __bf16* A, int lda, const __bf16* W, int ldw,
                               const float* bias, float* C, int ldc, int M, int N, int K, int acc)
{
    dim3 grid((N + 127) / 128, M / 64);
    gemm_bf16<<<grid, 128, 0, st>>>(A, lda, W, ldw, bias, C, ldc, M, N, K, acc);
}

extern "C" void kernel_launch(void* const* d_in, const int* in_sizes, int n_in,
                              void* d_out, int out_size, void* d_ws, size_t ws_size,
                              hipStream_t stream)
{
    const float* encoder_out = (const float*)d_in[0];
    const float* enc_att_w   = (const float*)d_in[1];
    const float* enc_att_b   = (const float*)d_in[2];
    const float* dec_att_w   = (const float*)d_in[3];
    const float* dec_att_b   = (const float*)d_in[4];
    const float* full_att_w  = (const float*)d_in[5];
    const float* full_att_b  = (const float*)d_in[6];
    const float* embed       = (const float*)d_in[7];
    const float* W_ih        = (const float*)d_in[8];
    const float* W_hh        = (const float*)d_in[9];
    const float* b_ih        = (const float*)d_in[10];
    const float* b_hh        = (const float*)d_in[11];
    const float* init_h_w    = (const float*)d_in[12];
    const float* init_h_b    = (const float*)d_in[13];
    const float* init_c_w    = (const float*)d_in[14];
    const float* init_c_b    = (const float*)d_in[15];
    const float* f_beta_w    = (const float*)d_in[16];
    const float* f_beta_b    = (const float*)d_in[17];
    const float* fc_w        = (const float*)d_in[18];
    const float* fc_b        = (const float*)d_in[19];
    const int*   enc_caps    = (const int*)d_in[20];
    const int*   cap_lengths = (const int*)d_in[21];

    float* out = (float*)d_out;
    float* out_preds   = out;                                   // B*T*V
    float* out_caps    = out + (size_t)BB * TT * VV;            // B*L
    float* out_declen  = out_caps + BB * LL;                    // B
    float* out_alphas  = out_declen + BB;                       // B*T*P
    float* out_sortind = out_alphas + (size_t)BB * TT * PP;     // B

    // ---- workspace carve-up ----
    char* w = (char*)d_ws;
    size_t off = 0;
    auto alloc = [&](size_t bytes) -> void* {
        void* p = w + off;
        off = (off + bytes + 255) & ~(size_t)255;
        return p;
    };
    int*    sort_ind   = (int*)    alloc(BB * sizeof(int));
    int*    dec_len    = (int*)    alloc(BB * sizeof(int));
    float*  att1       = (float*)  alloc((size_t)BB * PP * AD * 4);
    float*  att2       = (float*)  alloc((size_t)BB * AD * 4);
    float*  e_buf      = (float*)  alloc((size_t)BB * PP * 4);
    float*  alpha      = (float*)  alloc((size_t)BB * PP * 4);
    float*  awe        = (float*)  alloc((size_t)BB * ENCD * 4);
    float*  gate_lin   = (float*)  alloc((size_t)BB * ENCD * 4);
    float*  gates      = (float*)  alloc((size_t)BB * GD * 4);
    float*  h_buf      = (float*)  alloc((size_t)BB * HD * 4);
    float*  c_buf      = (float*)  alloc((size_t)BB * HD * 4);
    float*  preds_ws   = (float*)  alloc((size_t)BB * VV * 4);
    __bf16* enc_bf     = (__bf16*) alloc((size_t)BB * PP * ENCD * 2);   // 51.4 MB, L2 resident
    __bf16* mean_bf    = (__bf16*) alloc((size_t)BB * ENCD * 2);
    __bf16* h_bf       = (__bf16*) alloc((size_t)BB * HD * 2);
    __bf16* x_bf       = (__bf16*) alloc((size_t)BB * XD * 2);
    __bf16* enc_att_wb = (__bf16*) alloc((size_t)AD * ENCD * 2);
    __bf16* dec_att_wb = (__bf16*) alloc((size_t)AD * HD * 2);
    __bf16* f_beta_wb  = (__bf16*) alloc((size_t)ENCD * HD * 2);
    __bf16* W_ih_b     = (__bf16*) alloc((size_t)GD * XD * 2);
    __bf16* W_hh_b     = (__bf16*) alloc((size_t)GD * HD * 2);
    __bf16* fc_wb      = (__bf16*) alloc((size_t)VV * HD * 2);
    __bf16* init_h_wb  = (__bf16*) alloc((size_t)HD * ENCD * 2);
    __bf16* init_c_wb  = (__bf16*) alloc((size_t)HD * ENCD * 2);

    // ---- setup: sort, meta outputs, weight + activation conversion ----
    sort_kernel<<<1, BB, 0, stream>>>(cap_lengths, sort_ind, dec_len);
    meta_out_kernel<<<(BB * LL + 255) / 256, 256, 0, stream>>>(sort_ind, dec_len, enc_caps,
                                                               out_caps, out_declen, out_sortind);
    auto conv = [&](const float* s, __bf16* d, size_t n) {
        f2bf_kernel<<<(unsigned)((n + 255) / 256), 256, 0, stream>>>(s, d, n);
    };
    conv(enc_att_w, enc_att_wb, (size_t)AD * ENCD);
    conv(dec_att_w, dec_att_wb, (size_t)AD * HD);
    conv(f_beta_w,  f_beta_wb,  (size_t)ENCD * HD);
    conv(W_ih,      W_ih_b,     (size_t)GD * XD);
    conv(W_hh,      W_hh_b,     (size_t)GD * HD);
    conv(fc_w,      fc_wb,      (size_t)VV * HD);
    conv(init_h_w,  init_h_wb,  (size_t)HD * ENCD);
    conv(init_c_w,  init_c_wb,  (size_t)HD * ENCD);
    enc_gather_bf16_kernel<<<dim3(PP * ENCD / 256, BB), 256, 0, stream>>>(encoder_out, sort_ind, enc_bf);
    mean_kernel<<<dim3(ENCD / 256, BB), 256, 0, stream>>>(encoder_out, sort_ind, mean_bf);

    // ---- precompute: h0, c0, att1 ----
    launch_gemm(stream, mean_bf, ENCD, init_h_wb, ENCD, init_h_b, h_buf, HD, BB, HD, ENCD, 0);
    launch_gemm(stream, mean_bf, ENCD, init_c_wb, ENCD, init_c_b, c_buf, HD, BB, HD, ENCD, 0);
    f2bf_kernel<<<(BB * HD + 255) / 256, 256, 0, stream>>>(h_buf, h_bf, (size_t)BB * HD);
    // att1 = enc(sorted,bf16) @ enc_att_w^T + b
    launch_gemm(stream, enc_bf, ENCD, enc_att_wb, ENCD, enc_att_b, att1, AD, BB * PP, AD, ENCD, 0);

    // ---- decode loop ----
    for (int t = 0; t < TT; ++t) {
        // att2 = h @ dec_att_w^T + b
        launch_gemm(stream, h_bf, HD, dec_att_wb, HD, dec_att_b, att2, AD, BB, AD, HD, 0);
        // e = relu(att1 + att2) . full_att_w + b
        att_e_kernel<<<(BB * PP * 32 + 255) / 256, 256, 0, stream>>>(att1, att2, full_att_w,
                                                                     full_att_b, e_buf);
        // alpha = softmax(e); masked alphas -> out
        softmax_kernel<<<BB, 256, 0, stream>>>(e_buf, alpha, out_alphas, dec_len, t);
        // awe = alpha-weighted sum of encoder rows
        awe_kernel<<<dim3(ENCD / 256, BB), 256, 0, stream>>>(alpha, encoder_out, sort_ind, awe);
        // gate = sigmoid(h @ f_beta^T + b); x = [embed_t, gate*awe] in bf16
        launch_gemm(stream, h_bf, HD, f_beta_wb, HD, f_beta_b, gate_lin, ENCD, BB, ENCD, HD, 0);
        gate_x_kernel<<<dim3(XD / 256, BB), 256, 0, stream>>>(gate_lin, awe, embed, enc_caps,
                                                              sort_ind, t, x_bf);
        // gates = x @ W_ih^T + b_ih  (+)  h @ W_hh^T + b_hh
        launch_gemm(stream, x_bf, XD, W_ih_b, XD, b_ih, gates, GD, BB, GD, XD, 0);
        launch_gemm(stream, h_bf, HD, W_hh_b, HD, b_hh, gates, GD, BB, GD, HD, 1);
        // LSTM cell update (writes fp32 h/c and bf16 h)
        lstm_kernel<<<dim3(HD / 256, BB), 256, 0, stream>>>(gates, c_buf, h_buf, h_bf);
        // preds = h_new @ fc_w^T + fc_b, masked scatter to output
        launch_gemm(stream, h_bf, HD, fc_wb, HD, fc_b, preds_ws, VV, BB, VV, HD, 0);
        preds_out_kernel<<<dim3((VV + 255) / 256, BB), 256, 0, stream>>>(preds_ws, dec_len, t, out_preds);
    }
}
